// SparkFFN_14431090115239
// MI455X (gfx1250) — matrix operand order repaired and checked by reading, not compiled, not run on hardware
//
#include <hip/hip_runtime.h>
#include <math.h>

// Problem constants (match reference)
#define BB   4
#define SS   4096
#define DD   1024
#define FF   2048
#define MTOK (BB * SS)            // 16384 tokens
#define EPS_ 1e-6f
// norm_ppf(1 - 307/2048) = Phi^-1(0.85009765625)
#define COEFF 1.0368522f

typedef __attribute__((ext_vector_type(16))) __bf16       v16bf;
typedef __attribute__((ext_vector_type(8)))  float        v8f;
typedef __attribute__((ext_vector_type(4)))  unsigned int u32x4;
typedef int __attribute__((__vector_size__(16)))          i32x4;   // matches builtin param

#define AS1 __attribute__((address_space(1)))
#define AS3 __attribute__((address_space(3)))

// ---------------- async global->LDS (CDNA5) with guarded fallback ----------
#if defined(__has_builtin)
#  if __has_builtin(__builtin_amdgcn_global_load_async_to_lds_b128)
#    define HAS_ASYNC_LDS 1
#  endif
#endif
#ifndef HAS_ASYNC_LDS
#  define HAS_ASYNC_LDS 0
#endif

static __device__ __forceinline__ void cp_async16(const void* g, void* l) {
#if HAS_ASYNC_LDS
    __builtin_amdgcn_global_load_async_to_lds_b128(
        (AS1 i32x4*)g, (AS3 i32x4*)l, 0, 0);
#else
    *(u32x4*)l = *(const u32x4*)g;
#endif
}

#if HAS_ASYNC_LDS
#  if __has_builtin(__builtin_amdgcn_s_wait_asynccnt)
#    define WAIT_ASYNC(n) __builtin_amdgcn_s_wait_asynccnt(n)
#  else
#    define WAIT_ASYNC(n) asm volatile("s_wait_asynccnt %0" ::"n"(n))
#  endif
#else
#  define WAIT_ASYNC(n) ((void)0)
#endif

// ---- fp32 -> packed bf16x2 with round-to-nearest-even ----
static __device__ __forceinline__ unsigned f2bf_pk(float lo, float hi) {
    unsigned ul = __builtin_bit_cast(unsigned, lo);
    unsigned uh = __builtin_bit_cast(unsigned, hi);
    ul = (ul + (0x7FFFu + ((ul >> 16) & 1u))) >> 16;
    uh = (uh + (0x7FFFu + ((uh >> 16) & 1u))) >> 16;
    return (ul & 0xFFFFu) | (uh << 16);
}

// ---- build a 16-element bf16 fragment from two 16B LDS chunks ----
static __device__ __forceinline__ v16bf ld_frag(const unsigned* p0, const unsigned* p1) {
    union { u32x4 u[2]; v16bf v; } c;
    c.u[0] = *(const u32x4*)p0;
    c.u[1] = *(const u32x4*)p1;
    return c.v;
}

// ---- stage 16 fp32 from global into LDS as bf16 with fused relu(v-th) ----
static __device__ __forceinline__ void stage16_relu(const float* __restrict__ g, unsigned* s, float th) {
    const float4* p = (const float4*)g;
    float4 a0 = p[0], a1 = p[1], a2 = p[2], a3 = p[3];
    a0.x = fmaxf(a0.x - th, 0.f); a0.y = fmaxf(a0.y - th, 0.f);
    a0.z = fmaxf(a0.z - th, 0.f); a0.w = fmaxf(a0.w - th, 0.f);
    a1.x = fmaxf(a1.x - th, 0.f); a1.y = fmaxf(a1.y - th, 0.f);
    a1.z = fmaxf(a1.z - th, 0.f); a1.w = fmaxf(a1.w - th, 0.f);
    a2.x = fmaxf(a2.x - th, 0.f); a2.y = fmaxf(a2.y - th, 0.f);
    a2.z = fmaxf(a2.z - th, 0.f); a2.w = fmaxf(a2.w - th, 0.f);
    a3.x = fmaxf(a3.x - th, 0.f); a3.y = fmaxf(a3.y - th, 0.f);
    a3.z = fmaxf(a3.z - th, 0.f); a3.w = fmaxf(a3.w - th, 0.f);
    u32x4 lo = { f2bf_pk(a0.x, a0.y), f2bf_pk(a0.z, a0.w),
                 f2bf_pk(a1.x, a1.y), f2bf_pk(a1.z, a1.w) };
    u32x4 hi = { f2bf_pk(a2.x, a2.y), f2bf_pk(a2.z, a2.w),
                 f2bf_pk(a3.x, a3.y), f2bf_pk(a3.z, a3.w) };
    *(u32x4*)s       = lo;
    *((u32x4*)s + 1) = hi;
}

// LDS tile row = 32 bf16 = 16 uints, padded to 20 uints (80B, 16B aligned, conflict-avoiding)
#define ROWU 20

// ============================================================
// Pre-pass: fp32 -> bf16 (packed), 8 floats / thread
// ============================================================
__global__ __launch_bounds__(256) void cvt_bf16(const float* __restrict__ src,
                                                unsigned* __restrict__ dst,
                                                int n8) {
    int i = blockIdx.x * 256 + threadIdx.x;
    if (i >= n8) return;
    const float4* p = (const float4*)src + (size_t)i * 2;
    float4 a = p[0], b = p[1];
    u32x4 v = { f2bf_pk(a.x, a.y), f2bf_pk(a.z, a.w),
                f2bf_pk(b.x, b.y), f2bf_pk(b.z, b.w) };
    *((u32x4*)dst + i) = v;
}

// ============================================================
// GEMM1: h[t][f] = sum_d x[t][d] * w1[f][d]   (K = DD)
// bf16 operands, fully-async double-buffered LDS staging (zero staging VALU)
// ============================================================
__global__ __launch_bounds__(256) void spark_gemm1(const unsigned short* __restrict__ xb,
                                                   const unsigned short* __restrict__ w1b,
                                                   float* __restrict__ h) {
    __shared__ unsigned Xs[2][128 * ROWU];
    __shared__ unsigned Ws[2][128 * ROWU];

    const int tid   = threadIdx.x;
    const int lane  = tid & 31;
    const int wave  = tid >> 5;
    const int tBase = blockIdx.x * 128;   // token block
    const int fBase = blockIdx.y * 128;   // FF block

    // copy mapping: 4 threads per row, one 16B chunk each; two row-groups (0..63, 64..127)
    const int crow = tid >> 2;
    const int cq   = tid & 3;
    const char* gx0 = (const char*)(xb  + (size_t)(tBase + crow)      * DD) + cq * 16;
    const char* gx1 = (const char*)(xb  + (size_t)(tBase + crow + 64) * DD) + cq * 16;
    const char* gw0 = (const char*)(w1b + (size_t)(fBase + crow)      * DD) + cq * 16;
    const char* gw1 = (const char*)(w1b + (size_t)(fBase + crow + 64) * DD) + cq * 16;

    auto issue = [&](int buf, int k0) {
        const int bo = k0 * 2;  // byte offset along K
        cp_async16(gx0 + bo, (char*)&Xs[buf][ crow       * ROWU] + cq * 16);
        cp_async16(gx1 + bo, (char*)&Xs[buf][(crow + 64) * ROWU] + cq * 16);
        cp_async16(gw0 + bo, (char*)&Ws[buf][ crow       * ROWU] + cq * 16);
        cp_async16(gw1 + bo, (char*)&Ws[buf][(crow + 64) * ROWU] + cq * 16);
    };

    // wave -> 32x64 sub-tile of the 128x128 block
    const int waveM = (wave >> 1) * 32;
    const int waveN = (wave & 1) * 64;
    const int r  = lane & 15;
    const int hh = lane >> 4;

    v8f acc[2][4] = {};
    constexpr int NK = DD / 32;

    issue(0, 0);
    for (int i = 0; i < NK; ++i) {
        const int cur = i & 1;
        if (i + 1 < NK) {
            issue(cur ^ 1, (i + 1) * 32);
            WAIT_ASYNC(4);          // tile i complete (in-order), tile i+1 in flight
        } else {
            WAIT_ASYNC(0);
        }
        __syncthreads();

        v16bf afrag[2];
        v16bf bfrag[4];
#pragma unroll
        for (int mi = 0; mi < 2; ++mi) {
            const unsigned* ab = &Xs[cur][(waveM + mi * 16 + r) * ROWU];
            afrag[mi] = ld_frag(ab + hh * 4, ab + 8 + hh * 4);
        }
#pragma unroll
        for (int ni = 0; ni < 4; ++ni) {
            const unsigned* bb = &Ws[cur][(waveN + ni * 16 + r) * ROWU];
            bfrag[ni] = ld_frag(bb + hh * 4, bb + 8 + hh * 4);
        }
#pragma unroll
        for (int mi = 0; mi < 2; ++mi)
#pragma unroll
            for (int ni = 0; ni < 4; ++ni)
                acc[mi][ni] = __builtin_amdgcn_wmma_f32_16x16x32_bf16(
                    false, afrag[mi], false, bfrag[ni], (short)0, acc[mi][ni], false, false);

        __syncthreads();            // all reads of buf[cur] done before it is re-issued
    }

    // C/D layout: VGPR j -> (m = hh*8 + j, n = r); h stays fp32 (L2-resident, re-read twice)
#pragma unroll
    for (int mi = 0; mi < 2; ++mi) {
        const int tb = tBase + waveM + mi * 16 + hh * 8;
#pragma unroll
        for (int ni = 0; ni < 4; ++ni) {
            const int f = fBase + waveN + ni * 16 + r;
#pragma unroll
            for (int j = 0; j < 8; ++j)
                h[(size_t)(tb + j) * FF + f] = acc[mi][ni][j];
        }
    }
}

// ============================================================
// Stats: theta[t] = mean + max(std_unbiased, eps) * COEFF
// ============================================================
__global__ __launch_bounds__(256) void spark_stats(const float* __restrict__ h,
                                                   float* __restrict__ theta) {
    const int t = blockIdx.x;
    const int tid = threadIdx.x;
    const float* row = h + (size_t)t * FF;

    float s = 0.f, q = 0.f;
#pragma unroll
    for (int i = 0; i < FF / 256; ++i) {
        float v = row[tid + i * 256];
        s += v;
        q += v * v;
    }
    __shared__ float ssum[256];
    __shared__ float ssq[256];
    ssum[tid] = s;
    ssq[tid]  = q;
    __syncthreads();
    for (int off = 128; off > 0; off >>= 1) {
        if (tid < off) {
            ssum[tid] += ssum[tid + off];
            ssq[tid]  += ssq[tid + off];
        }
        __syncthreads();
    }
    if (tid == 0) {
        float mu  = ssum[0] / (float)FF;
        float var = (ssq[0] - (float)FF * mu * mu) / (float)(FF - 1);
        float sg  = sqrtf(fmaxf(var, 0.f));
        sg = fmaxf(sg, EPS_);
        theta[t] = mu + sg * COEFF;
    }
}

// ============================================================
// GEMM2: out[t][n] = sum_f relu(h[t][f]-theta[t]) * w2[n][f]   (K = FF)
// A: fp32 h with fused clamp+cvt (VALU); B: bf16 w2 via async copy
// ============================================================
__global__ __launch_bounds__(256) void spark_gemm2(const float* __restrict__ h,
                                                   const float* __restrict__ theta,
                                                   const unsigned short* __restrict__ w2b,
                                                   float* __restrict__ out) {
    __shared__ unsigned Hs[128 * ROWU];
    __shared__ unsigned Ws2[128 * ROWU];

    const int tid   = threadIdx.x;
    const int lane  = tid & 31;
    const int wave  = tid >> 5;
    const int tBase = blockIdx.x * 128;   // token block
    const int nBase = blockIdx.y * 128;   // D block

    // A-tile staging (fp32 h): 2 threads per row, 16 floats each
    const int lrow  = tid >> 1;
    const int lhalf = tid & 1;
    const float* hg = h + (size_t)(tBase + lrow) * FF + lhalf * 16;
    unsigned* hs = &Hs[lrow * ROWU + lhalf * 8];
    const float th = theta[tBase + lrow];

    // B-tile async copy mapping (bf16 w2): 4 threads per row, 16B chunks
    const int crow = tid >> 2;
    const int cq   = tid & 3;
    const char* gw0 = (const char*)(w2b + (size_t)(nBase + crow)      * FF) + cq * 16;
    const char* gw1 = (const char*)(w2b + (size_t)(nBase + crow + 64) * FF) + cq * 16;
    char* lw0 = (char*)&Ws2[ crow       * ROWU] + cq * 16;
    char* lw1 = (char*)&Ws2[(crow + 64) * ROWU] + cq * 16;

    const int waveM = (wave >> 1) * 32;
    const int waveN = (wave & 1) * 64;
    const int r  = lane & 15;
    const int hh = lane >> 4;

    v8f acc[2][4] = {};

    for (int k0 = 0; k0 < FF; k0 += 32) {
        __syncthreads();                    // previous compute done with LDS
        cp_async16(gw0 + k0 * 2, lw0);      // DMA B tile while VALU stages A tile
        cp_async16(gw1 + k0 * 2, lw1);
        stage16_relu(hg + k0, hs, th);
        if (k0 + 32 < FF) __builtin_prefetch(hg + k0 + 32, 0, 0);
        WAIT_ASYNC(0);
        __syncthreads();

        v16bf afrag[2];
        v16bf bfrag[4];
#pragma unroll
        for (int mi = 0; mi < 2; ++mi) {
            const unsigned* ab = &Hs[(waveM + mi * 16 + r) * ROWU];
            afrag[mi] = ld_frag(ab + hh * 4, ab + 8 + hh * 4);
        }
#pragma unroll
        for (int ni = 0; ni < 4; ++ni) {
            const unsigned* bb = &Ws2[(waveN + ni * 16 + r) * ROWU];
            bfrag[ni] = ld_frag(bb + hh * 4, bb + 8 + hh * 4);
        }
#pragma unroll
        for (int mi = 0; mi < 2; ++mi)
#pragma unroll
            for (int ni = 0; ni < 4; ++ni)
                acc[mi][ni] = __builtin_amdgcn_wmma_f32_16x16x32_bf16(
                    false, afrag[mi], false, bfrag[ni], (short)0, acc[mi][ni], false, false);
    }

#pragma unroll
    for (int mi = 0; mi < 2; ++mi) {
        const int tb = tBase + waveM + mi * 16 + hh * 8;
#pragma unroll
        for (int ni = 0; ni < 4; ++ni) {
            const int n = nBase + waveN + ni * 16 + r;
#pragma unroll
            for (int j = 0; j < 8; ++j)
                out[(size_t)(tb + j) * DD + n] = acc[mi][ni][j];
        }
    }
}

// ============================================================
extern "C" void kernel_launch(void* const* d_in, const int* in_sizes, int n_in,
                              void* d_out, int out_size, void* d_ws, size_t ws_size,
                              hipStream_t stream) {
    (void)in_sizes; (void)n_in; (void)out_size; (void)ws_size;
    const float* x  = (const float*)d_in[0];
    const float* w1 = (const float*)d_in[1];
    const float* w2 = (const float*)d_in[2];
    float* out = (float*)d_out;

    // workspace layout
    char* ws = (char*)d_ws;
    float* h_ws  = (float*)ws;                                   // 128 MB fp32
    ws += (size_t)MTOK * FF * sizeof(float);
    float* theta = (float*)ws;                                   // 64 KB
    ws += (size_t)MTOK * sizeof(float);
    unsigned short* xb  = (unsigned short*)ws;                   // 32 MB bf16
    ws += (size_t)MTOK * DD * sizeof(unsigned short);
    unsigned short* w1b = (unsigned short*)ws;                   // 4 MB bf16
    ws += (size_t)FF * DD * sizeof(unsigned short);
    unsigned short* w2b = (unsigned short*)ws;                   // 4 MB bf16

    dim3 blk(256);
    const int nx8 = MTOK * DD / 8, nw8 = FF * DD / 8;
    cvt_bf16<<<dim3(nx8 / 256), blk, 0, stream>>>(x,  (unsigned*)xb,  nx8);
    cvt_bf16<<<dim3(nw8 / 256), blk, 0, stream>>>(w1, (unsigned*)w1b, nw8);
    cvt_bf16<<<dim3(nw8 / 256), blk, 0, stream>>>(w2, (unsigned*)w2b, nw8);

    spark_gemm1<<<dim3(MTOK / 128, FF / 128), blk, 0, stream>>>(xb, w1b, h_ws);
    spark_stats<<<dim3(MTOK), blk, 0, stream>>>(h_ws, theta);
    spark_gemm2<<<dim3(MTOK / 128, DD / 128), blk, 0, stream>>>(h_ws, theta, w2b, out);
}
